// Consistency_10574209482833
// MI455X (gfx1250) — matrix operand order repaired and checked by reading, not compiled, not run on hardware
//
#include <hip/hip_runtime.h>
#include <hip/hip_bf16.h>
#include <cstdint>
#include <cstddef>

typedef __bf16 bf16_t;
typedef __attribute__((ext_vector_type(16))) __bf16 v16bf;
typedef __attribute__((ext_vector_type(8)))  float  v8f;

#define DD    256   // D_EDGE == D_SINK
#define ROWS  32    // rows of MLP output we ever need (sizes <= N_MAX = 32)
#define EMAX  256
#define LA_LD 257   // padded row stride to avoid 64-bank conflicts on column pass

__device__ inline v8f v8f_zero() {
  v8f z;
#pragma unroll
  for (int i = 0; i < 8; ++i) z[i] = 0.f;
  return z;
}

__device__ inline float wred_max(float v) {
#pragma unroll
  for (int o = 16; o > 0; o >>= 1) v = fmaxf(v, __shfl_xor(v, o, 32));
  return v;
}
__device__ inline float wred_sum(float v) {
#pragma unroll
  for (int o = 16; o > 0; o >>= 1) v += __shfl_xor(v, o, 32);
  return v;
}
__device__ inline int clamp33(int v) { return v < 0 ? 0 : (v > 32 ? 32 : v); }

// ---- WMMA fragment loaders (contiguous k = 16h + s in A and B, wave32) -------------
// A 16x32: lane<16 -> M=lane, K=koff..koff+15 with koff=0;
//          lane>=16 -> same M, koff=16.
__device__ inline v16bf load_a_lds(const bf16_t* X, int mr, int kb, int lane) {
  const int row  = mr + (lane & 15);
  const int koff = (lane & 16) ? 16 : 0;
  const bf16_t* p = X + row * DD + kb + koff;
  v16bf a;
#pragma unroll
  for (int t = 0; t < 8; ++t) { a[t] = p[t]; a[8 + t] = p[8 + t]; }
  return a;
}
// B 32x16 (K x N) from pre-transposed bf16 weights Wt[n][k]:
// lane<16 -> N=lane, K=kb..kb+15 ; lane>=16 -> N=lane-16, K=kb+16..kb+31.
// One contiguous 32-byte (v16bf) load per lane -> 2x global_load_b128.
__device__ inline v16bf load_b_t(const bf16_t* __restrict__ Wt, int kb, int nb, int lane) {
  const int col = nb + (lane & 15);
  const int k0  = kb + ((lane & 16) ? 16 : 0);
  return *(const v16bf*)(Wt + (size_t)col * DD + k0);
}

// ---- setup: recover per-graph node/edge offsets deterministically ----------
__global__ void init_offsets(int* node_first, int* edge_first, int* edge_count, int ng) {
  int i = blockIdx.x * blockDim.x + threadIdx.x;
  if (i < ng) { node_first[i] = 0x7fffffff; edge_first[i] = 0x7fffffff; edge_count[i] = 0; }
}
__global__ void node_scan(const int* __restrict__ graph_idx, int* node_first, int n) {
  int i = blockIdx.x * blockDim.x + threadIdx.x;
  if (i < n) atomicMin(&node_first[graph_idx[i]], i);
}
__global__ void edge_scan(const int* __restrict__ from_idx, const int* __restrict__ graph_idx,
                          int* edge_first, int* edge_count, int n) {
  int e = blockIdx.x * blockDim.x + threadIdx.x;
  if (e < n) {
    int g = graph_idx[from_idx[e]];
    atomicMin(&edge_first[g], e);
    atomicAdd(&edge_count[g], 1);
  }
}
// Transpose + convert weights f32[k][n] -> bf16 Wt[n][k] (one-time, 128 KB)
__global__ void convert_weights(const float* __restrict__ W, bf16_t* __restrict__ Wt, int n) {
  int idx = blockIdx.x * blockDim.x + threadIdx.x;
  if (idx < n) {
    int k = idx >> 8, c = idx & 255;
    Wt[(size_t)c * DD + k] = (bf16_t)W[idx];
  }
}

// ---- MLP: Y = relu(X@W1 + b1)@W2 + b2 on the 32 surviving rows per graph ---
// One workgroup (8 waves) per graph. Each wave owns 32 output columns.
__global__ __launch_bounds__(256) void mlp_kernel(
    const float* __restrict__ feats, const bf16_t* __restrict__ Wt1, const float* __restrict__ b1,
    const bf16_t* __restrict__ Wt2, const float* __restrict__ b2,
    const int* __restrict__ edge_first, const int* __restrict__ edge_count,
    const int* __restrict__ qsz, const int* __restrict__ csz,
    float* __restrict__ qc) {
  __shared__ bf16_t Xs[ROWS * DD];
  __shared__ bf16_t Hs[ROWS * DD];
  const int g = blockIdx.x;
  const int tid = threadIdx.x, lane = tid & 31, wave = tid >> 5;
  const int ef = edge_first[g], ec = edge_count[g];

  for (int idx = tid; idx < ROWS * DD; idx += 256) {
    int r = idx >> 8;
    float v = (r < ec) ? feats[(size_t)(ef + r) * DD + (idx & 255)] : 0.f;
    Xs[idx] = (bf16_t)v;
  }
  __syncthreads();

  const int nb = wave * 32;
  const int colb = nb + (lane & 15);
  const int rofs = (lane & 16) ? 8 : 0;

  // ---- layer 1 ----
  v8f a00 = v8f_zero(), a01 = v8f_zero(), a10 = v8f_zero(), a11 = v8f_zero();
  for (int kb = 0; kb < DD; kb += 32) {
    if (kb + 32 < DD) __builtin_prefetch(&Wt1[(size_t)colb * DD + kb + 32], 0, 0);
    v16bf fa0 = load_a_lds(Xs, 0, kb, lane);
    v16bf fa1 = load_a_lds(Xs, 16, kb, lane);
    v16bf fb0 = load_b_t(Wt1, kb, nb, lane);
    v16bf fb1 = load_b_t(Wt1, kb, nb + 16, lane);
    a00 = __builtin_amdgcn_wmma_f32_16x16x32_bf16(false, fa0, false, fb0, (short)0, a00, false, false);
    a01 = __builtin_amdgcn_wmma_f32_16x16x32_bf16(false, fa0, false, fb1, (short)0, a01, false, false);
    a10 = __builtin_amdgcn_wmma_f32_16x16x32_bf16(false, fa1, false, fb0, (short)0, a10, false, false);
    a11 = __builtin_amdgcn_wmma_f32_16x16x32_bf16(false, fa1, false, fb1, (short)0, a11, false, false);
  }
  {
    const float bia0 = b1[colb], bia1 = b1[colb + 16];
#pragma unroll
    for (int v = 0; v < 8; ++v) {
      const int r0 = v + rofs;
      Hs[r0 * DD + colb]             = (bf16_t)fmaxf(a00[v] + bia0, 0.f);
      Hs[r0 * DD + colb + 16]        = (bf16_t)fmaxf(a01[v] + bia1, 0.f);
      Hs[(16 + r0) * DD + colb]      = (bf16_t)fmaxf(a10[v] + bia0, 0.f);
      Hs[(16 + r0) * DD + colb + 16] = (bf16_t)fmaxf(a11[v] + bia1, 0.f);
    }
  }
  __syncthreads();

  // ---- layer 2 ----
  a00 = v8f_zero(); a01 = v8f_zero(); a10 = v8f_zero(); a11 = v8f_zero();
  for (int kb = 0; kb < DD; kb += 32) {
    if (kb + 32 < DD) __builtin_prefetch(&Wt2[(size_t)colb * DD + kb + 32], 0, 0);
    v16bf fa0 = load_a_lds(Hs, 0, kb, lane);
    v16bf fa1 = load_a_lds(Hs, 16, kb, lane);
    v16bf fb0 = load_b_t(Wt2, kb, nb, lane);
    v16bf fb1 = load_b_t(Wt2, kb, nb + 16, lane);
    a00 = __builtin_amdgcn_wmma_f32_16x16x32_bf16(false, fa0, false, fb0, (short)0, a00, false, false);
    a01 = __builtin_amdgcn_wmma_f32_16x16x32_bf16(false, fa0, false, fb1, (short)0, a01, false, false);
    a10 = __builtin_amdgcn_wmma_f32_16x16x32_bf16(false, fa1, false, fb0, (short)0, a10, false, false);
    a11 = __builtin_amdgcn_wmma_f32_16x16x32_bf16(false, fa1, false, fb1, (short)0, a11, false, false);
  }
  {
    const int size = (g & 1) ? csz[g >> 1] : qsz[g >> 1];
    const float bia0 = b2[colb], bia1 = b2[colb + 16];
    float* outg = qc + (size_t)g * ROWS * DD;
#pragma unroll
    for (int v = 0; v < 8; ++v) {
      const int r0 = v + rofs, r1 = 16 + v + rofs;
      float y00 = (r0 < size) ? a00[v] + bia0 : 0.f;
      float y01 = (r0 < size) ? a01[v] + bia1 : 0.f;
      float y10 = (r1 < size) ? a10[v] + bia0 : 0.f;
      float y11 = (r1 < size) ? a11[v] + bia1 : 0.f;
      outg[r0 * DD + colb]      = y00;
      outg[r0 * DD + colb + 16] = y01;
      outg[r1 * DD + colb]      = y10;
      outg[r1 * DD + colb + 16] = y11;
    }
  }
}

// ---- fused Sinkhorn (20 iters, fully LDS-resident) + score -----------------
// LDS layout (dynamic, 16B-aligned regions):
//   la[256*257] f32 | T[33*33] (padded) | fq/tq/fc/tc[256] | cf[32*256] f32 | red[64]
#define LA_BYTES   (EMAX * LA_LD * 4)   // 263168 (16B aligned)
#define T_BYTES_P  4368                 // 33*33*4 = 4356, padded to 16B
#define IDX_BYTES  (4 * EMAX * 4)       // 4096
#define CF_OFF     (LA_BYTES + T_BYTES_P + IDX_BYTES)  // 271632, 16B aligned
#define CF_BYTES   (ROWS * DD * 4)      // 32768
#define RED_OFF    (CF_OFF + CF_BYTES)
#define SMEM_TOTAL (RED_OFF + 256)

__global__ __launch_bounds__(256) void sinkhorn_kernel(
    const float* __restrict__ Tin, const int* __restrict__ from_idx, const int* __restrict__ to_idx,
    const int* __restrict__ node_first, const int* __restrict__ edge_first,
    const int* __restrict__ edge_count, const float* __restrict__ qc,
    float* __restrict__ out) {
  extern __shared__ char smem[];
  float* la = (float*)smem;
  float* Ts = (float*)(smem + LA_BYTES);
  int*   fq = (int*)(smem + LA_BYTES + T_BYTES_P);
  int*   tq = fq + EMAX;
  int*   fc = tq + EMAX;
  int*   tc = fc + EMAX;
  float* cf = (float*)(smem + CF_OFF);
  float* red = (float*)(smem + RED_OFF);

  const int b = blockIdx.x;
  const int tid = threadIdx.x, lane = tid & 31, wave = tid >> 5;
  const int gq = 2 * b, gc = 2 * b + 1;

  // Kick off async DMA of c_feat (32 KB) into LDS; it lands while the 20
  // Sinkhorn iterations run out of LDS. Tracked by ASYNCcnt.
  {
    const unsigned base_lds = __builtin_amdgcn_groupstaticsize() + (unsigned)CF_OFF;
    const float* gsrc = qc + (size_t)gc * ROWS * DD;
    for (int idx = tid; idx < (ROWS * DD) / 4; idx += 256) {
      unsigned ldst = base_lds + (unsigned)idx * 16u;
      unsigned long long ga = (unsigned long long)(uintptr_t)(gsrc + idx * 4);
      asm volatile("global_load_async_to_lds_b128 %0, %1, off"
                   :: "v"(ldst), "v"(ga) : "memory");
    }
  }

  // zero-padded transport plan (33x33)
  for (int idx = tid; idx < 33 * 33; idx += 256) {
    int r = idx / 33, c = idx - r * 33;
    Ts[idx] = (r < 32 && c < 32) ? Tin[(size_t)b * 1024 + r * 32 + c] : 0.f;
  }
  // per-edge local node indices (pad value = 32 -> hits the zero pad of Ts)
  {
    const int efq = edge_first[gq], nq = node_first[gq];
    const int efc = edge_first[gc], nc = node_first[gc];
    const int ecq = min(edge_count[gq], EMAX), ecc = min(edge_count[gc], EMAX);
    int vfq = 32, vtq = 32, vfc = 32, vtc = 32;
    if (tid < ecq) {
      vfq = clamp33(from_idx[efq + tid] - nq);
      vtq = clamp33(to_idx[efq + tid] - nq);
    }
    if (tid < ecc) {
      vfc = clamp33(from_idx[efc + tid] - nc);
      vtc = clamp33(to_idx[efc + tid] - nc);
    }
    fq[tid] = vfq; tq[tid] = vtq; fc[tid] = vfc; tc[tid] = vtc;
  }
  __syncthreads();

  // la = (straight + cross) / TEMPERATURE
  for (int idx = tid; idx < EMAX * EMAX; idx += 256) {
    const int i = idx >> 8, j = idx & 255;
    const float s = Ts[fq[i] * 33 + fc[j]] * Ts[tq[i] * 33 + tc[j]];
    const float x = Ts[fq[i] * 33 + tc[j]] * Ts[tq[i] * 33 + fc[j]];
    la[i * LA_LD + j] = (s + x) * 10.0f;  // 1/0.1
  }
  __syncthreads();

  // 20 Sinkhorn iterations, one wave per row/column, all in LDS
  for (int it = 0; it < 20; ++it) {
    for (int i = wave; i < EMAX; i += 8) {
      float* row = la + i * LA_LD;
      float v[8]; float m = -3.4e38f;
#pragma unroll
      for (int t = 0; t < 8; ++t) { v[t] = row[lane + 32 * t]; m = fmaxf(m, v[t]); }
      m = wred_max(m);
      float s = 0.f;
#pragma unroll
      for (int t = 0; t < 8; ++t) s += __expf(v[t] - m);
      s = wred_sum(s);
      const float lse = m + __logf(s);
#pragma unroll
      for (int t = 0; t < 8; ++t) row[lane + 32 * t] = v[t] - lse;
    }
    __syncthreads();
    for (int j = wave; j < EMAX; j += 8) {
      float v[8]; float m = -3.4e38f;
#pragma unroll
      for (int t = 0; t < 8; ++t) { v[t] = la[(lane + 32 * t) * LA_LD + j]; m = fmaxf(m, v[t]); }
      m = wred_max(m);
      float s = 0.f;
#pragma unroll
      for (int t = 0; t < 8; ++t) s += __expf(v[t] - m);
      s = wred_sum(s);
      const float lse = m + __logf(s);
#pragma unroll
      for (int t = 0; t < 8; ++t) la[(lane + 32 * t) * LA_LD + j] = v[t] - lse;
    }
    __syncthreads();
  }

  // plan = exp(la)
  for (int idx = tid; idx < EMAX * EMAX; idx += 256) {
    const int i = idx >> 8, j = idx & 255;
    la[i * LA_LD + j] = __expf(la[i * LA_LD + j]);
  }
  // c_feat async copy must be complete before reading cf
  asm volatile("s_wait_asynccnt 0x0" ::: "memory");
  __syncthreads();

  // score = -sum relu(q - plan @ c); c rows >= 32 are zero, so k < 32 suffices
  float tot = 0.f;
  const float* qrow = qc + (size_t)gq * ROWS * DD;
  const int d = tid;
  for (int i = 0; i < EMAX; ++i) {
    float acc = 0.f;
    const float* pr = la + i * LA_LD;
#pragma unroll 8
    for (int k = 0; k < ROWS; ++k) acc = fmaf(pr[k], cf[k * DD + d], acc);
    const float qv = (i < ROWS) ? qrow[i * DD + d] : 0.f;
    const float diff = qv - acc;
    tot += diff > 0.f ? diff : 0.f;
  }
  tot = wred_sum(tot);
  if (lane == 0) red[wave] = tot;
  __syncthreads();
  if (tid == 0) {
    float s = 0.f;
#pragma unroll
    for (int w = 0; w < 8; ++w) s += red[w];
    out[b] = -s;
  }
}

extern "C" void kernel_launch(void* const* d_in, const int* in_sizes, int n_in,
                              void* d_out, int out_size, void* d_ws, size_t ws_size,
                              hipStream_t stream) {
  const float* feats = (const float*)d_in[0];
  const float* Tin   = (const float*)d_in[1];
  const float* W1    = (const float*)d_in[2];
  const float* b1    = (const float*)d_in[3];
  const float* W2    = (const float*)d_in[4];
  const float* b2    = (const float*)d_in[5];
  const int* from_idx  = (const int*)d_in[6];
  const int* to_idx    = (const int*)d_in[7];
  const int* graph_idx = (const int*)d_in[8];
  const int* qsz       = (const int*)d_in[9];
  const int* csz       = (const int*)d_in[10];
  float* out = (float*)d_out;

  const int Bb = in_sizes[9];          // 256
  const int ng = 2 * Bb;               // 512
  const int n_nodes = in_sizes[8];     // 12288
  const int n_edges = in_sizes[6];     // 102400

  char* ws = (char*)d_ws;
  float* qc = (float*)ws;                                    // ng*32*256 f32 = 16 MB
  size_t qc_bytes = (size_t)ng * ROWS * DD * sizeof(float);
  int* node_first = (int*)(ws + qc_bytes);
  int* edge_first = node_first + ng;
  int* edge_count = edge_first + ng;
  size_t wt_off = qc_bytes + 8192;                           // room for 3*ng ints, 16B aligned
  bf16_t* Wt1 = (bf16_t*)(ws + wt_off);                      // 128 KB
  bf16_t* Wt2 = (bf16_t*)(ws + wt_off + (size_t)DD * DD * sizeof(bf16_t));

  init_offsets<<<(ng + 255) / 256, 256, 0, stream>>>(node_first, edge_first, edge_count, ng);
  node_scan<<<(n_nodes + 255) / 256, 256, 0, stream>>>(graph_idx, node_first, n_nodes);
  edge_scan<<<(n_edges + 255) / 256, 256, 0, stream>>>(from_idx, graph_idx, edge_first, edge_count, n_edges);
  convert_weights<<<(DD * DD + 255) / 256, 256, 0, stream>>>(W1, Wt1, DD * DD);
  convert_weights<<<(DD * DD + 255) / 256, 256, 0, stream>>>(W2, Wt2, DD * DD);
  mlp_kernel<<<ng, 256, 0, stream>>>(feats, Wt1, b1, Wt2, b2, edge_first, edge_count, qsz, csz, qc);
  sinkhorn_kernel<<<Bb, 256, SMEM_TOTAL, stream>>>(Tin, from_idx, to_idx, node_first,
                                                   edge_first, edge_count, qc, out);
}
